// HopfieldMemory_57123065036824
// MI455X (gfx1250) — matrix operand order repaired and checked by reading, not compiled, not run on hardware
//
#include <hip/hip_runtime.h>
#include <hip/hip_bf16.h>
#include <math.h>

typedef __bf16 bf16t;
typedef __attribute__((ext_vector_type(16))) __bf16 v16bf;
typedef __attribute__((ext_vector_type(8)))  __bf16 v8bf;
typedef __attribute__((ext_vector_type(8)))  float  v8f;

constexpr int kS    = 4096;
constexpr int kH    = 2048;
constexpr int kBS   = 16384;   // B*S rows
constexpr int kNCOL = 384;     // q|k|v concatenated output cols
constexpr int kCH   = 64;      // chunk size
// nc == N_SLOTS == 64: slot pointer never wraps; active-slot mask == (slot < chunk)

__device__ __forceinline__ v8f wmma_bf16(v16bf a, v16bf b, v8f c) {
  return __builtin_amdgcn_wmma_f32_16x16x32_bf16(false, a, false, b, (short)0, c, false, false);
}

__device__ __forceinline__ v16bf pack16(v8bf a, v8bf b) {
  v16bf r;
#pragma unroll
  for (int i = 0; i < 8; ++i) { r[i] = a[i]; r[i + 8] = b[i]; }
  return r;
}

// A fragment (16 rows x 32 K, bf16) from row-major bf16: lane L holds row L%16,
// K = {h*8..h*8+7} U {16+h*8..16+h*8+7}, h = L/16 (ISA 16-bit A layout). ld multiple of 8.
__device__ __forceinline__ v16bf fragA_bf(const bf16t* base, int ld, int lane) {
  const bf16t* p = base + (size_t)(lane & 15) * ld + ((lane >> 4) << 3);
  v8bf a = *(const v8bf*)p;
  v8bf b = *(const v8bf*)(p + 16);
  return pack16(a, b);
}

// Same A fragment but converting f32 -> bf16 on the fly.
__device__ __forceinline__ v16bf fragA_f32(const float* base, int ld, int lane) {
  const float* p = base + (size_t)(lane & 15) * ld + ((lane >> 4) << 3);
  v16bf r;
#pragma unroll
  for (int i = 0; i < 8; ++i) { r[i] = (bf16t)p[i]; r[i + 8] = (bf16t)p[i + 16]; }
  return r;
}

// B fragment (32 K x 16 N): weights stored row-major (N x K); lane L holds col N=L%16,
// K = {h*8..h*8+7} U {16+h*8..16+h*8+7}, h = L/16 (matches A fragment order).
__device__ __forceinline__ v16bf fragB_bf(const bf16t* base, int ld, int lane) {
  const bf16t* p = base + (size_t)(lane & 15) * ld + ((lane >> 4) << 3);
  v8bf a = *(const v8bf*)p;
  v8bf b = *(const v8bf*)(p + 16);
  return pack16(a, b);
}

// C layout: VGPR i -> row i (lanes 0-15) / row i+8 (lanes 16-31), col = lane%16.
__device__ __forceinline__ void store_tileC(float* dst, int ld, v8f c, int lane) {
  int n = lane & 15, rh = (lane >> 4) << 3;
#pragma unroll
  for (int i = 0; i < 8; ++i) dst[(size_t)(rh + i) * ld + n] = c[i];
}

// ---------------- weight conversion ----------------
__global__ void __launch_bounds__(256, 1)
prep_k(const float* __restrict__ Wq, const float* __restrict__ Wk,
       const float* __restrict__ Wv, const float* __restrict__ Wo,
       const float* __restrict__ Wg,
       bf16t* __restrict__ Wcatb, bf16t* __restrict__ wgb,
       bf16t* __restrict__ Woutb) {
  size_t i = (size_t)blockIdx.x * 256 + threadIdx.x;
  const size_t NQ = (size_t)128 * kH, NWO = (size_t)kH * 128;
  if (i < NQ)            Wcatb[i] = (bf16t)Wq[i];
  else if (i < 2 * NQ)   Wcatb[i] = (bf16t)Wk[i - NQ];
  else if (i < 3 * NQ)   Wcatb[i] = (bf16t)Wv[i - 2 * NQ];
  else {
    size_t j = i - 3 * NQ;
    if (j < NWO)         Woutb[j] = (bf16t)Wo[j];
    else { j -= NWO; if (j < (size_t)kH) wgb[j] = (bf16t)Wg[j]; }
  }
}

// G = W_out^T W_out (128x128), for ||out_row|| = sqrt(r^T G r)
__global__ void __launch_bounds__(128, 1)
gram_k(const float* __restrict__ Wo, bf16t* __restrict__ Gb) {
  int i = blockIdx.x, j = threadIdx.x;
  float a = 0.f;
  for (int h = 0; h < kH; ++h) a += Wo[(size_t)h * 128 + i] * Wo[(size_t)h * 128 + j];
  Gb[(size_t)i * 128 + j] = (bf16t)a;
}

// ---------------- fused QKV + gate projection: (16384 x 2048) x (384 x 2048)^T ----------------
__global__ void __launch_bounds__(256, 1)
qkvg_gemm_k(const float* __restrict__ x, const bf16t* __restrict__ Wcat,
            const bf16t* __restrict__ wgb, float* __restrict__ qkv,
            float* __restrict__ glog) {
  __shared__ bf16t xs[64][40];               // 64 rows x 32 K, ld 40 (16B-aligned, conflict-free)
  const int tid = threadIdx.x, lane = tid & 31, wave = tid >> 5;
  const int wm = wave & 1, wn = wave >> 1;   // 2(M) x 4(N) waves
  const int rowbase = blockIdx.x * 64;
  v8f acc[2][6] = {};
  float gacc = 0.f;
  for (int kb = 0; kb < kH; kb += 32) {
    {
      int r = tid >> 2, seg = tid & 3;
      const float* p = x + (size_t)(rowbase + r) * kH + kb + seg * 8;
      float4 a = *(const float4*)p;
      float4 b = *(const float4*)(p + 4);
      if (kb + 32 < kH) __builtin_prefetch(p + 32, 0, 0);
      bf16t* d = &xs[r][seg * 8];
      d[0] = (bf16t)a.x; d[1] = (bf16t)a.y; d[2] = (bf16t)a.z; d[3] = (bf16t)a.w;
      d[4] = (bf16t)b.x; d[5] = (bf16t)b.y; d[6] = (bf16t)b.z; d[7] = (bf16t)b.w;
    }
    __syncthreads();
    if (tid < 64) {                          // gate logit partial (reuses bf16 x tile)
#pragma unroll
      for (int k2 = 0; k2 < 32; ++k2)
        gacc += (float)xs[tid][k2] * (float)wgb[kb + k2];
    }
    v16bf bfrag[6];
#pragma unroll
    for (int nt = 0; nt < 6; ++nt)
      bfrag[nt] = fragB_bf(Wcat + (size_t)(wn * 96 + nt * 16) * kH + kb, kH, lane);
#pragma unroll
    for (int mt = 0; mt < 2; ++mt) {
      v16bf af = fragA_bf(&xs[wm * 32 + mt * 16][0], 40, lane);
#pragma unroll
      for (int nt = 0; nt < 6; ++nt)
        acc[mt][nt] = wmma_bf16(af, bfrag[nt], acc[mt][nt]);
    }
    __syncthreads();
  }
#pragma unroll
  for (int mt = 0; mt < 2; ++mt)
#pragma unroll
    for (int nt = 0; nt < 6; ++nt)
      store_tileC(qkv + (size_t)(rowbase + wm * 32 + mt * 16) * kNCOL + wn * 96 + nt * 16,
                  kNCOL, acc[mt][nt], lane);
  if (tid < 64) glog[rowbase + tid] = gacc;
}

// ---------------- per-chunk gated aggregation (all 64 chunks in parallel) ----------------
__device__ __forceinline__ float reduce128(float* red, float* bcast, int tid) {
  __syncthreads();
  for (int s2 = 64; s2 > 0; s2 >>= 1) { if (tid < s2) red[tid] += red[tid + s2]; __syncthreads(); }
  if (tid == 0) *bcast = red[0];
  __syncthreads();
  return *bcast;
}

__global__ void __launch_bounds__(256, 1)
chunk_agg_k(const float* __restrict__ qkv, const float* __restrict__ glog,
            const float* __restrict__ gate_b,
            bf16t* __restrict__ Kaggb, bf16t* __restrict__ VaggTb,
            float* __restrict__ wsarr) {
  const int t = blockIdx.x, tid = threadIdx.x;          // 256 threads = (b, c) pairs
  __shared__ float gate_s[256], ink_s[256], inv_s[256];
  __shared__ float kacc[2][128], vacc[2][128];
  __shared__ float red[128], bc[3];
  {
    int b = tid >> 6, c = tid & 63;
    size_t rg = (size_t)b * kS + (size_t)t * kCH + c;
    const float* kp = qkv + rg * kNCOL + 128;
    const float* vp = qkv + rg * kNCOL + 256;
    float nk = 0.f, nv = 0.f;
    for (int d = 0; d < 128; ++d) { float a = kp[d]; nk += a * a; float e = vp[d]; nv += e * e; }
    ink_s[tid] = 1.f / fmaxf(sqrtf(nk), 1e-12f);
    inv_s[tid] = 1.f / fmaxf(sqrtf(nv), 1e-12f);
    gate_s[tid] = 1.f / (1.f + __expf(-(glog[rg] + gate_b[0])));
  }
  __syncthreads();
  {
    int g = tid >> 7, j = tid & 127;
    float ka = 0.f, va = 0.f;
    for (int r = 0; r < 128; ++r) {
      int idx = (g << 7) + r;
      int b = idx >> 6, c = idx & 63;
      size_t rg = (size_t)b * kS + (size_t)t * kCH + c;
      float w = gate_s[idx];
      ka += w * ink_s[idx] * qkv[rg * kNCOL + 128 + j];
      va += w * inv_s[idx] * qkv[rg * kNCOL + 256 + j];
    }
    kacc[g][j] = ka; vacc[g][j] = va;
  }
  __syncthreads();
  if (tid < 128) red[tid] = gate_s[tid] + gate_s[tid + 128];
  float gs = reduce128(red, &bc[0], tid);
  if (tid == 0) wsarr[t] = gs * (1.f / 256.f);          // write_str = mean gate
  if (tid < 128) { float v = kacc[0][tid] + kacc[1][tid]; kacc[0][tid] = v; red[tid] = v * v; }
  float kn = reduce128(red, &bc[1], tid);
  if (tid < 128) { float v = vacc[0][tid] + vacc[1][tid]; vacc[0][tid] = v; red[tid] = v * v; }
  float vn = reduce128(red, &bc[2], tid);
  if (tid < 128) {
    float iks = 1.f / fmaxf(sqrtf(kn), 1e-12f);
    float ivs = 1.f / fmaxf(sqrtf(vn), 1e-12f);
    Kaggb[(size_t)t * 128 + tid] = (bf16t)(kacc[0][tid] * iks);  // (slot, dim) row-major
    VaggTb[(size_t)tid * 64 + t] = (bf16t)(vacc[0][tid] * ivs);  // (dim, slot) row-major
  }
}

// ---------------- causal chunk attention: scores -> softmax -> r ----------------
__global__ void __launch_bounds__(256, 1)
attn_read_k(const float* __restrict__ qkv, const bf16t* __restrict__ Kaggb,
            const bf16t* __restrict__ VaggTb, const float* __restrict__ log_beta,
            float* __restrict__ rbuf) {
  const int t = blockIdx.x, hh = blockIdx.y;            // chunk, row-half
  const int tid = threadIdx.x, lane = tid & 31, wave = tid >> 5;
  __shared__ float sc[128][68];
  __shared__ bf16t at[128][72];
  if (t == 0) {                                          // no active slots -> r = 0 (ref zeroes it)
    if (tid < 128) {
      int idx = hh * 128 + tid; int b = idx >> 6, c = idx & 63;
      float* p = rbuf + ((size_t)b * kS + c) * 128;
      for (int d = 0; d < 128; ++d) p[d] = 0.f;
    }
    return;
  }
  const float beta = __expf(log_beta[0]);
  const int mt = wave;                                   // one 16-row tile per wave
  int idx0 = hh * 128 + mt * 16; int bb = idx0 >> 6, c0 = idx0 & 63;
  const float* abase = qkv + ((size_t)bb * kS + (size_t)t * kCH + c0) * kNCOL;
  {
    v8f acc[4] = {};
#pragma unroll
    for (int ks = 0; ks < 4; ++ks) {
      v16bf af = fragA_f32(abase + ks * 32, kNCOL, lane);
#pragma unroll
      for (int nt = 0; nt < 4; ++nt)
        acc[nt] = wmma_bf16(af, fragB_bf(Kaggb + (size_t)nt * 16 * 128 + ks * 32, 128, lane),
                            acc[nt]);
    }
    int n = lane & 15, rh = (lane >> 4) << 3;
#pragma unroll
    for (int nt = 0; nt < 4; ++nt) {
      int col = nt * 16 + n;
      float m = (col >= t) ? -1e9f : 0.f;                // inactive-slot mask (same as ref -1e9)
#pragma unroll
      for (int i = 0; i < 8; ++i)
        sc[mt * 16 + rh + i][col] = acc[nt][i] * beta + m;
    }
  }
  __syncthreads();
  if (tid < 128) {                                       // row softmax over 64 slots
    float mx = -3.0e38f;
    for (int j = 0; j < 64; ++j) mx = fmaxf(mx, sc[tid][j]);
    float sum = 0.f;
    for (int j = 0; j < 64; ++j) sum += __expf(sc[tid][j] - mx);
    float inv = 1.f / sum;
    for (int j = 0; j < 64; ++j) at[tid][j] = (bf16t)(__expf(sc[tid][j] - mx) * inv);
  }
  __syncthreads();
  {
    v8f acc[8] = {};
#pragma unroll
    for (int ks = 0; ks < 2; ++ks) {
      v16bf af = fragA_bf(&at[mt * 16][ks * 32], 72, lane);
#pragma unroll
      for (int nt = 0; nt < 8; ++nt)
        acc[nt] = wmma_bf16(af, fragB_bf(VaggTb + (size_t)nt * 16 * 64 + ks * 32, 64, lane),
                            acc[nt]);
    }
    float* cb = rbuf + ((size_t)bb * kS + (size_t)t * kCH + c0) * 128;
#pragma unroll
    for (int nt = 0; nt < 8; ++nt)
      store_tileC(cb + nt * 16, 128, acc[nt], lane);
  }
}

// ---------------- generic K=128 GEMM: C(f32) = A(16384x128) x B(N x 128)^T ----------------
template <int NT>
__global__ void __launch_bounds__(256, 1)
gemm_rs_k(const void* __restrict__ Araw, int a_is_f32,
          const bf16t* __restrict__ Bw, float* __restrict__ C, int ldc) {
  __shared__ bf16t As[64][136];
  const int tid = threadIdx.x, lane = tid & 31, wave = tid >> 5;
  const int wm = wave & 1, wn = wave >> 1;               // 2(M) x 4(N)
  const int rowbase = blockIdx.x * 64;
  const int colbase = blockIdx.y * (NT * 64);
  {
    int r = tid >> 2, seg = tid & 3;
    if (a_is_f32) {
      const float* p = (const float*)Araw + (size_t)(rowbase + r) * 128 + seg * 32;
#pragma unroll
      for (int i = 0; i < 32; i += 4) {
        float4 v = *(const float4*)(p + i);
        bf16t* d = &As[r][seg * 32 + i];
        d[0] = (bf16t)v.x; d[1] = (bf16t)v.y; d[2] = (bf16t)v.z; d[3] = (bf16t)v.w;
      }
    } else {
      const bf16t* p = (const bf16t*)Araw + (size_t)(rowbase + r) * 128 + seg * 32;
#pragma unroll
      for (int i = 0; i < 32; ++i) As[r][seg * 32 + i] = p[i];
    }
  }
  __syncthreads();
  v8f acc[2][NT] = {};
#pragma unroll
  for (int ks = 0; ks < 4; ++ks) {
    v16bf bfrag[NT];
#pragma unroll
    for (int nt = 0; nt < NT; ++nt)
      bfrag[nt] = fragB_bf(Bw + (size_t)(colbase + (wn * NT + nt) * 16) * 128 + ks * 32, 128, lane);
#pragma unroll
    for (int mt = 0; mt < 2; ++mt) {
      v16bf af = fragA_bf(&As[wm * 32 + mt * 16][ks * 32], 136, lane);
#pragma unroll
      for (int nt = 0; nt < NT; ++nt)
        acc[mt][nt] = wmma_bf16(af, bfrag[nt], acc[mt][nt]);
    }
  }
#pragma unroll
  for (int mt = 0; mt < 2; ++mt)
#pragma unroll
    for (int nt = 0; nt < NT; ++nt)
      store_tileC(C + (size_t)(rowbase + wm * 32 + mt * 16) * ldc + colbase + (wn * NT + nt) * 16,
                  ldc, acc[mt][nt], lane);
}

// ---------------- clamp scale: s = min(10/sqrt(r.u), 1);  rs = bf16(r*s) ----------------
__global__ void __launch_bounds__(256, 1)
clamp_scale_k(const float* __restrict__ rbuf, const float* __restrict__ ubuf,
              bf16t* __restrict__ rsb) {
  int lane = threadIdx.x & 31, w = threadIdx.x >> 5;
  int warp = blockIdx.x * 8 + w;                         // 64 blocks * 8 waves = 512 warps
  for (int row = warp; row < kBS; row += 512) {
    float4 u4 = *(const float4*)(ubuf + (size_t)row * 128 + lane * 4);
    float4 r4 = *(const float4*)(rbuf + (size_t)row * 128 + lane * 4);
    float p = u4.x * r4.x + u4.y * r4.y + u4.z * r4.z + u4.w * r4.w;
#pragma unroll
    for (int off = 16; off; off >>= 1) p += __shfl_xor(p, off, 32);  // wave32 reduction
    float nrm = sqrtf(fmaxf(p, 0.f));
    float s = fminf(10.f / fmaxf(nrm, 1e-6f), 1.f);
    bf16t* d = rsb + (size_t)row * 128 + lane * 4;
    d[0] = (bf16t)(r4.x * s); d[1] = (bf16t)(r4.y * s);
    d[2] = (bf16t)(r4.z * s); d[3] = (bf16t)(r4.w * s);
  }
}

__global__ void __launch_bounds__(64, 1)
meanws_k(const float* __restrict__ wsarr, float* __restrict__ out) {
  if (threadIdx.x == 0) {
    float s = 0.f;
    for (int i = 0; i < 64; ++i) s += wsarr[i];
    *out = s * (1.f / 64.f);
  }
}

extern "C" void kernel_launch(void* const* d_in, const int* in_sizes, int n_in,
                              void* d_out, int out_size, void* d_ws, size_t ws_size,
                              hipStream_t stream) {
  (void)in_sizes; (void)n_in; (void)out_size; (void)ws_size;
  const float* x  = (const float*)d_in[0];
  const float* Wq = (const float*)d_in[1];
  const float* Wk = (const float*)d_in[2];
  const float* Wv = (const float*)d_in[3];
  const float* Wo = (const float*)d_in[4];
  const float* Wg = (const float*)d_in[5];
  const float* gb = (const float*)d_in[6];
  const float* lb = (const float*)d_in[7];

  char* ws = (char*)d_ws;
  size_t o = 0;
  auto take = [&](size_t bytes) { char* p = ws + o; o += (bytes + 255) & ~(size_t)255; return p; };
  bf16t* Wcatb  = (bf16t*)take((size_t)kNCOL * kH * sizeof(bf16t));
  bf16t* wgb    = (bf16t*)take((size_t)kH * sizeof(bf16t));
  bf16t* Woutb  = (bf16t*)take((size_t)kH * 128 * sizeof(bf16t));
  bf16t* Gb     = (bf16t*)take((size_t)128 * 128 * sizeof(bf16t));
  float* qkv    = (float*)take((size_t)kBS * kNCOL * sizeof(float));
  float* glog   = (float*)take((size_t)kBS * sizeof(float));
  bf16t* Kaggb  = (bf16t*)take((size_t)64 * 128 * sizeof(bf16t));
  bf16t* VaggTb = (bf16t*)take((size_t)128 * 64 * sizeof(bf16t));
  float* wsarr  = (float*)take((size_t)64 * sizeof(float));
  float* rbuf   = (float*)take((size_t)kBS * 128 * sizeof(float));
  float* ubuf   = (float*)take((size_t)kBS * 128 * sizeof(float));
  bf16t* rsb    = (bf16t*)take((size_t)kBS * 128 * sizeof(bf16t));

  float* outp = (float*)d_out;

  prep_k<<<4104, 256, 0, stream>>>(Wq, Wk, Wv, Wo, Wg, Wcatb, wgb, Woutb);
  gram_k<<<128, 128, 0, stream>>>(Wo, Gb);
  qkvg_gemm_k<<<kBS / 64, 256, 0, stream>>>(x, Wcatb, wgb, qkv, glog);
  chunk_agg_k<<<64, 256, 0, stream>>>(qkv, glog, gb, Kaggb, VaggTb, wsarr);
  attn_read_k<<<dim3(64, 2), 256, 0, stream>>>(qkv, Kaggb, VaggTb, lb, rbuf);
  gemm_rs_k<2><<<dim3(kBS / 64, 1), 256, 0, stream>>>(rbuf, 1, Gb, ubuf, 128);
  clamp_scale_k<<<64, 256, 0, stream>>>(rbuf, ubuf, rsb);
  gemm_rs_k<4><<<dim3(kBS / 64, 8), 256, 0, stream>>>(rsb, 0, Woutb, outp, kH);
  meanws_k<<<1, 64, 0, stream>>>(wsarr, outp + (size_t)kBS * kH);
}
